// E2EVRPHeteroGNN_2010044695108
// MI455X (gfx1250) — matrix operand order repaired and checked by reading, not compiled, not run on hardware
//
#include <hip/hip_runtime.h>

typedef __attribute__((ext_vector_type(16))) _Float16 v16h;
typedef __attribute__((ext_vector_type(8)))  float    v8f;

// ---------------------------------------------------------------- utilities
__global__ void k_zero(float* __restrict__ p, size_t n) {
    for (size_t i = (size_t)blockIdx.x * blockDim.x + threadIdx.x; i < n;
         i += (size_t)gridDim.x * blockDim.x)
        p[i] = 0.0f;
}

// ------------------------------------------------- weight pre-swizzle into WMMA B-fragment order
// Input:  Wl, Wr each [16][64x64] f32 (16 = L*8+ri pairs)
// Output: out[m][4096] f16, m = pair*2 + which(0=Wl,1=Wr)
// Fragment layout: o = ((kc*4 + n)*32 + lane)*16 + e  <->  W[(kc*32 + (lane>=16?16:0) + e)*64 + n*16 + (lane&15)]
__global__ void k_wswizzle(const float* __restrict__ Wl, const float* __restrict__ Wr,
                           _Float16* __restrict__ out) {
    int t = blockIdx.x * 256 + threadIdx.x;          // 32 * 4096 total
    if (t >= 32 * 4096) return;
    const int m = t >> 12;
    const int o = t & 4095;
    const int pair = m >> 1, which = m & 1;
    const int kc = o >> 11;
    const int n = (o >> 9) & 3;
    const int lane = (o >> 4) & 31;
    const int e = o & 15;
    const int k = kc * 32 + ((lane >> 4) ? 16 : 0) + e;
    const int col = n * 16 + (lane & 15);
    const float* W = which ? Wr : Wl;
    out[(size_t)m * 4096 + o] = (_Float16)W[(size_t)pair * 4096 + k * 64 + col];
}

// ------------------------------------------------- input projection: relu(raw@Win+b)
__global__ void k_inproj(const float* __restrict__ raw, const float* __restrict__ W,
                         const float* __restrict__ b, float* __restrict__ x, int n) {
    int t = blockIdx.x * blockDim.x + threadIdx.x;
    int row = t >> 6, col = t & 63;
    if (row >= n) return;
    const float* rp = raw + (size_t)row * 10;
    float a = b[col];
#pragma unroll
    for (int k = 0; k < 10; ++k) a += rp[k] * W[k * 64 + col];
    x[(size_t)row * 64 + col] = fmaxf(a, 0.0f);
}

// ------------------------------------------------- edge scatter: msum[dst]+=x[src], cnt[dst]+=1
__global__ void k_scatter(const float* __restrict__ xsrc, const int* __restrict__ src,
                          const int* __restrict__ dst, float* __restrict__ msum,
                          float* __restrict__ cnt, int ne) {
    long long t = (long long)blockIdx.x * blockDim.x + threadIdx.x;
    int e = (int)(t >> 4);
    int c4 = ((int)t & 15) * 4;
    if (e >= ne) return;
    int s = src[e], d = dst[e];
    const float4 v = *(const float4*)(xsrc + (size_t)s * 64 + c4);
    float* mp = msum + (size_t)d * 64 + c4;
    atomicAdd(mp + 0, v.x);
    atomicAdd(mp + 1, v.y);
    atomicAdd(mp + 2, v.z);
    atomicAdd(mp + 3, v.w);
    if (c4 == 0) atomicAdd(&cnt[d], 1.0f);
}

// ------------------------------------------------- SAGE: out = mean@Wl + bl + xdst@Wr, L2-norm rows
// Ws = pre-swizzled f16 fragments: [0..4095]=Wl, [4096..8191]=Wr.
__launch_bounds__(256)
__global__ void k_sage_gemm(const float* __restrict__ msum, const float* __restrict__ cnt,
                            const float* __restrict__ xdst,
                            const _Float16* __restrict__ Ws, const float* __restrict__ bl,
                            float* __restrict__ nbuf, int ndst, int accum) {
    __shared__ _Float16 sW[8192];                    // 16 KB: swizzled Wl | Wr
    {
        // async global->LDS bulk copy: 256 threads x 4 x 16B = 16 KB, ASYNCcnt-tracked
        unsigned ldsa = (unsigned)(uintptr_t)(&sW[0]) + threadIdx.x * 16u;
        unsigned long long ga = (unsigned long long)(uintptr_t)Ws + threadIdx.x * 16ull;
        asm volatile(
            "global_load_async_to_lds_b128 %0, %1, off\n\t"
            "global_load_async_to_lds_b128 %0, %1, off offset:4096\n\t"
            "global_load_async_to_lds_b128 %0, %1, off offset:8192\n\t"
            "global_load_async_to_lds_b128 %0, %1, off offset:12288\n\t"
            "s_wait_asynccnt 0x0"
            :: "v"(ldsa), "v"(ga) : "memory");
    }
    __syncthreads();

    const int lane = threadIdx.x & 31;
    const int wave = threadIdx.x >> 5;
    const int row0 = (blockIdx.x * 8 + wave) * 16;
    if (row0 >= ndst) return;                        // wave-uniform: EXEC all-ones for WMMA

    const int r  = lane & 15;
    const int hi = lane >> 4;
    const int h8 = hi ? 16 : 0;
    const int rowA  = row0 + r;
    const bool va   = rowA < ndst;
    const size_t ra = (size_t)(va ? rowA : 0) * 64;
    const float rc  = va ? (1.0f / fmaxf(cnt[rowA], 1.0f)) : 0.0f;
    const float xs  = va ? 1.0f : 0.0f;
    const float* mrow = msum + ra;
    const float* xrow = xdst + ra;

    const v16h* fWl = (const v16h*)(sW);
    const v16h* fWr = (const v16h*)(sW + 4096);

    v8f acc[4];
#pragma unroll
    for (int n = 0; n < 4; ++n)
#pragma unroll
        for (int v = 0; v < 8; ++v) acc[n][v] = 0.0f;

#pragma unroll
    for (int kc = 0; kc < 2; ++kc) {
        const int k0 = kc * 32;
        // A fragment: elements 0..15 <-> K = k0+h8+0..15
        const float* mp = mrow + k0 + h8;
        const float* xp = xrow + k0 + h8;
        float4 m0 = *(const float4*)(mp);      float4 m1 = *(const float4*)(mp + 4);
        float4 m2 = *(const float4*)(mp + 8); float4 m3 = *(const float4*)(mp + 12);
        float4 x0 = *(const float4*)(xp);      float4 x1 = *(const float4*)(xp + 4);
        float4 x2 = *(const float4*)(xp + 8); float4 x3 = *(const float4*)(xp + 12);
        const float tm[16] = { m0.x, m0.y, m0.z, m0.w, m1.x, m1.y, m1.z, m1.w,
                               m2.x, m2.y, m2.z, m2.w, m3.x, m3.y, m3.z, m3.w };
        const float tx[16] = { x0.x, x0.y, x0.z, x0.w, x1.x, x1.y, x1.z, x1.w,
                               x2.x, x2.y, x2.z, x2.w, x3.x, x3.y, x3.z, x3.w };
        v16h aM, aX;
#pragma unroll
        for (int e = 0; e < 16; ++e) {
            aM[e] = (_Float16)(tm[e] * rc);
            aX[e] = (_Float16)(tx[e] * xs);
        }
        // hoist all 8 B fragments for this K-chunk, then issue the WMMA burst
        v16h bLa[4], bRa[4];
#pragma unroll
        for (int n = 0; n < 4; ++n) {
            const int f = (kc * 4 + n) * 32 + lane;  // contiguous 32B fragment per lane
            bLa[n] = fWl[f];
            bRa[n] = fWr[f];
        }
#pragma unroll
        for (int n = 0; n < 4; ++n) {
            acc[n] = __builtin_amdgcn_wmma_f32_16x16x32_f16(false, aM, false, bLa[n],
                                                            (short)0, acc[n], false, false);
            acc[n] = __builtin_amdgcn_wmma_f32_16x16x32_f16(false, aX, false, bRa[n],
                                                            (short)0, acc[n], false, false);
        }
    }

    // + bias
#pragma unroll
    for (int n = 0; n < 4; ++n) {
        const float bb = bl[n * 16 + r];
#pragma unroll
        for (int v = 0; v < 8; ++v) acc[n][v] += bb;
    }
    // L2-normalize each output row (row M = v + 8*hi spans one 16-lane half)
#pragma unroll
    for (int v = 0; v < 8; ++v) {
        float s = 0.0f;
#pragma unroll
        for (int n = 0; n < 4; ++n) s += acc[n][v] * acc[n][v];
        s += __shfl_xor(s, 1, 32);
        s += __shfl_xor(s, 2, 32);
        s += __shfl_xor(s, 4, 32);
        s += __shfl_xor(s, 8, 32);
        const float inv = 1.0f / fmaxf(sqrtf(s), 1e-12f);
#pragma unroll
        for (int n = 0; n < 4; ++n) acc[n][v] *= inv;
    }
    // store / accumulate (single owner per element -> plain RMW)
#pragma unroll
    for (int v = 0; v < 8; ++v) {
        const int grow = row0 + v + (hi ? 8 : 0);
        if (grow < ndst) {
#pragma unroll
            for (int n = 0; n < 4; ++n) {
                const size_t idx = (size_t)grow * 64 + n * 16 + r;
                nbuf[idx] = accum ? (nbuf[idx] + acc[n][v]) : acc[n][v];
            }
        }
    }
}

// ------------------------------------------------- layernorm(new + x) -> x (in place)
__global__ void k_ln(const float* __restrict__ nw, float* __restrict__ x,
                     const float* __restrict__ g, const float* __restrict__ b, int n) {
    int row = blockIdx.x * blockDim.x + threadIdx.x;
    if (row >= n) return;
    const float* np = nw + (size_t)row * 64;
    float* xp = x + (size_t)row * 64;
    float v[64];
    float m = 0.0f;
#pragma unroll
    for (int k = 0; k < 64; ++k) { v[k] = np[k] + xp[k]; m += v[k]; }
    m *= (1.0f / 64.0f);
    float var = 0.0f;
#pragma unroll
    for (int k = 0; k < 64; ++k) { float d = v[k] - m; var += d * d; }
    var *= (1.0f / 64.0f);
    const float inv = rsqrtf(var + 1e-5f);
#pragma unroll
    for (int k = 0; k < 64; ++k) xp[k] = (v[k] - m) * inv * g[k] + b[k];
}

// ------------------------------------------------- attention readout
__global__ void k_att1(const float* __restrict__ cust, const float* __restrict__ wa,
                       const float* __restrict__ ba, float* __restrict__ logits,
                       float* __restrict__ partmax, int n) {
    __shared__ float sm[256];
    int row = blockIdx.x * 256 + threadIdx.x;
    float d = -3.402823466e38f;
    if (row < n) {
        const float* cp = cust + (size_t)row * 64;
        float a = ba[0];
#pragma unroll
        for (int k = 0; k < 64; ++k) a += cp[k] * wa[k];
        logits[row] = a;
        d = a;
    }
    sm[threadIdx.x] = d;
    __syncthreads();
    for (int s = 128; s > 0; s >>= 1) {
        if (threadIdx.x < s) sm[threadIdx.x] = fmaxf(sm[threadIdx.x], sm[threadIdx.x + s]);
        __syncthreads();
    }
    if (threadIdx.x == 0) partmax[blockIdx.x] = sm[0];
}

__global__ void k_att2(const float* __restrict__ partmax, int nblk, float* __restrict__ gmax) {
    __shared__ float sm[256];
    float d = -3.402823466e38f;
    for (int i = threadIdx.x; i < nblk; i += 256) d = fmaxf(d, partmax[i]);
    sm[threadIdx.x] = d;
    __syncthreads();
    for (int s = 128; s > 0; s >>= 1) {
        if (threadIdx.x < s) sm[threadIdx.x] = fmaxf(sm[threadIdx.x], sm[threadIdx.x + s]);
        __syncthreads();
    }
    if (threadIdx.x == 0) gmax[0] = sm[0];
}

__global__ void k_att3(const float* __restrict__ cust, const float* __restrict__ logits,
                       const float* __restrict__ gmax, float* __restrict__ gacc,
                       float* __restrict__ Z, int n) {
    __shared__ float sacc[64];
    __shared__ float ssum[256];
    if (threadIdx.x < 64) sacc[threadIdx.x] = 0.0f;
    __syncthreads();
    int row = blockIdx.x * 256 + threadIdx.x;
    float e = 0.0f;
    if (row < n) {
        e = expf(logits[row] - gmax[0]);
        const float* cp = cust + (size_t)row * 64;
#pragma unroll
        for (int k = 0; k < 64; ++k) atomicAdd(&sacc[k], e * cp[k]);
    }
    ssum[threadIdx.x] = e;
    __syncthreads();
    for (int s = 128; s > 0; s >>= 1) {
        if (threadIdx.x < s) ssum[threadIdx.x] += ssum[threadIdx.x + s];
        __syncthreads();
    }
    if (threadIdx.x == 0) atomicAdd(Z, ssum[0]);
    if (threadIdx.x < 64) atomicAdd(&gacc[threadIdx.x], sacc[threadIdx.x]);
}

// ------------------------------------------------- head: scalars MLP + concat + 2-layer MLP
__global__ void k_final(const float* __restrict__ gacc, const float* __restrict__ Z,
                        const float* __restrict__ scal,
                        const float* __restrict__ Ws1, const float* __restrict__ bs1,
                        const float* __restrict__ Ws2, const float* __restrict__ bs2,
                        const float* __restrict__ Wo1, const float* __restrict__ bo1,
                        const float* __restrict__ Wo2, const float* __restrict__ bo2,
                        float* __restrict__ out) {
    __shared__ float comb[128];
    __shared__ float h1[64];
    __shared__ float h2[128];
    const int t = threadIdx.x;  // 128 threads
    if (t < 64) {
        comb[t] = gacc[t] / Z[0];
        float a = bs1[t];
#pragma unroll
        for (int i = 0; i < 5; ++i) a += scal[i] * Ws1[i * 64 + t];
        h1[t] = fmaxf(a, 0.0f);
    }
    __syncthreads();
    if (t < 64) {
        float a = bs2[t];
        for (int l = 0; l < 64; ++l) a += h1[l] * Ws2[l * 64 + t];
        comb[64 + t] = a;
    }
    __syncthreads();
    {
        float a = bo1[t];
        for (int j = 0; j < 128; ++j) a += comb[j] * Wo1[j * 128 + t];
        h2[t] = fmaxf(a, 0.0f);
    }
    __syncthreads();
    {
        float a = bo2[t];
        for (int j = 0; j < 128; ++j) a += h2[j] * Wo2[j * 128 + t];
        out[t] = a;
    }
}

// =====================================================================================
extern "C" void kernel_launch(void* const* d_in, const int* in_sizes, int n_in,
                              void* d_out, int out_size, void* d_ws, size_t ws_size,
                              hipStream_t stream) {
    (void)n_in; (void)out_size; (void)ws_size;
    // ---- inputs (setup_inputs dict order)
    const float* x_depot     = (const float*)d_in[0];
    const float* x_satellite = (const float*)d_in[1];
    const float* x_customer  = (const float*)d_in[2];
    const float* x_recharge  = (const float*)d_in[3];
    const float* scalars     = (const float*)d_in[4];
    const int* e_src[8], * e_dst[8];
    int ne[8];
    for (int i = 0; i < 8; ++i) {
        e_src[i] = (const int*)d_in[5 + 2 * i];
        e_dst[i] = (const int*)d_in[6 + 2 * i];
        ne[i]    = in_sizes[5 + 2 * i];
    }
    const float* Win  = (const float*)d_in[21];
    const float* b_in = (const float*)d_in[22];
    const float* Wl   = (const float*)d_in[23];
    const float* bl   = (const float*)d_in[24];
    const float* Wr   = (const float*)d_in[25];
    const float* ln_g = (const float*)d_in[26];
    const float* ln_b = (const float*)d_in[27];
    const float* wa   = (const float*)d_in[28];
    const float* ba   = (const float*)d_in[29];
    const float* Ws1  = (const float*)d_in[30];
    const float* bs1  = (const float*)d_in[31];
    const float* Ws2  = (const float*)d_in[32];
    const float* bs2  = (const float*)d_in[33];
    const float* Wo1  = (const float*)d_in[34];
    const float* bo1  = (const float*)d_in[35];
    const float* Wo2  = (const float*)d_in[36];
    const float* bo2  = (const float*)d_in[37];

    const int ND = in_sizes[0] / 10, NS = in_sizes[1] / 10;
    const int NC = in_sizes[2] / 10, NR = in_sizes[3] / 10;

    // ---- workspace carve-out (floats, 256B aligned)
    float* ws = (float*)d_ws;
    size_t off = 0;
    auto take = [&](size_t nf) { float* p = ws + off; off += (nf + 63) & ~(size_t)63; return p; };
    float* xC   = take((size_t)NC * 64);
    float* xS   = take((size_t)NS * 64);
    float* xR   = take((size_t)NR * 64);
    float* xD   = take(64);
    float* newC = take((size_t)NC * 64);
    float* newS = take((size_t)NS * 64);
    float* newR = take((size_t)NR * 64);
    float* msum = take((size_t)NC * 64);   // reused per relation (max dst = Customer)
    float* cnt  = take((size_t)NC);
    float* logits  = take((size_t)NC);
    float* partmax = take(2048);
    float* smallb  = take(128);            // [0..63]=gacc, [64]=Z, [65]=gmax
    _Float16* wsw  = (_Float16*)take(32 * 4096 / 2);  // swizzled weights: 32 matrices x 4096 f16

    // ---- 0) swizzle all Wl/Wr matrices into WMMA fragment order (f16)
    k_wswizzle<<<(32 * 4096) / 256, 256, 0, stream>>>(Wl, Wr, wsw);

    // ---- 1) input projections (NTYPES order: Depot, Satellite, Customer, Recharging)
    k_inproj<<<(ND * 64 + 255) / 256, 256, 0, stream>>>(x_depot,     Win + 0 * 640, b_in + 0 * 64, xD, ND);
    k_inproj<<<(NS * 64 + 255) / 256, 256, 0, stream>>>(x_satellite, Win + 1 * 640, b_in + 1 * 64, xS, NS);
    k_inproj<<<(NC * 64 + 255) / 256, 256, 0, stream>>>(x_customer,  Win + 2 * 640, b_in + 2 * 64, xC, NC);
    k_inproj<<<(NR * 64 + 255) / 256, 256, 0, stream>>>(x_recharge,  Win + 3 * 640, b_in + 3 * 64, xR, NR);

    // relation tables (RELS order)
    const float* xsrc_t[8] = { xC, xC, xS, xC, xD, xS, xC, xR };
    const float* xdst_t[8] = { xC, xS, xC, xC, xR, xR, xR, xR };
    float*       nd_t[8]   = { newC, newS, newC, newC, newR, newR, newR, newR };
    const int    ndst[8]   = { NC, NS, NC, NC, NR, NR, NR, NR };
    const int    accum[8]  = { 0, 0, 1, 1, 0, 1, 1, 1 };  // first writer per dst type clears

    auto zgrid = [](size_t n) { size_t b = (n + 255) / 256; return (int)(b < 8192 ? b : 8192); };

    // ---- 2) two hetero-SAGE layers
    for (int L = 0; L < 2; ++L) {
        for (int ri = 0; ri < 8; ++ri) {
            const int nd = ndst[ri];
            k_zero<<<zgrid((size_t)nd * 64), 256, 0, stream>>>(msum, (size_t)nd * 64);
            k_zero<<<zgrid((size_t)nd), 256, 0, stream>>>(cnt, (size_t)nd);
            const long long nth = (long long)ne[ri] * 16;
            k_scatter<<<(int)((nth + 255) / 256), 256, 0, stream>>>(
                xsrc_t[ri], e_src[ri], e_dst[ri], msum, cnt, ne[ri]);
            const int rowTiles = (nd + 15) / 16;
            const int blocks   = (rowTiles + 7) / 8;
            const _Float16* Wsp = wsw + (size_t)(L * 8 + ri) * 8192;
            const float* blp = bl + ((size_t)L * 8 + ri) * 64;
            k_sage_gemm<<<blocks, 256, 0, stream>>>(msum, cnt, xdst_t[ri],
                                                    Wsp, blp, nd_t[ri], nd, accum[ri]);
        }
        // layernorm (Depot never a destination -> skipped)
        k_ln<<<(NS + 255) / 256, 256, 0, stream>>>(newS, xS, ln_g + (L * 4 + 1) * 64, ln_b + (L * 4 + 1) * 64, NS);
        k_ln<<<(NC + 255) / 256, 256, 0, stream>>>(newC, xC, ln_g + (L * 4 + 2) * 64, ln_b + (L * 4 + 2) * 64, NC);
        k_ln<<<(NR + 255) / 256, 256, 0, stream>>>(newR, xR, ln_g + (L * 4 + 3) * 64, ln_b + (L * 4 + 3) * 64, NR);
    }

    // ---- 3) attention readout over customers + head MLP
    const int nblkA = (NC + 255) / 256;
    k_att1<<<nblkA, 256, 0, stream>>>(xC, wa, ba, logits, partmax, NC);
    k_att2<<<1, 256, 0, stream>>>(partmax, nblkA, smallb + 65);
    k_zero<<<1, 256, 0, stream>>>(smallb, 65);  // gacc[64] + Z
    k_att3<<<nblkA, 256, 0, stream>>>(xC, logits, smallb + 65, smallb, smallb + 64, NC);
    k_final<<<1, 128, 0, stream>>>(smallb, smallb + 64, scalars,
                                   Ws1, bs1, Ws2, bs2, Wo1, bo1, Wo2, bo2, (float*)d_out);
}
